// GraphModel_48490180772611
// MI455X (gfx1250) — matrix operand order repaired and checked by reading, not compiled, not run on hardware
//
#include <hip/hip_runtime.h>

#define N_NODES 50000
#define N_EDGES 800000

typedef __attribute__((ext_vector_type(16))) __bf16 v16bf;
typedef __attribute__((ext_vector_type(8)))  float  v8f;

union BFrag { v16bf v; unsigned short s[16]; uint4 q[2]; };

__device__ __forceinline__ unsigned short f2bf_rne(float f) {
  union { float f; unsigned int u; } x; x.f = f;
  unsigned int u = x.u;
  u += 0x7fffu + ((u >> 16) & 1u);   // round to nearest even
  return (unsigned short)(u >> 16);
}

// -------- pack weights W[K x NOUT] (f32 row-major) into WMMA B-fragment bf16 --------
// B 32x16 bf16 k order used here (contiguous, same as the A loads): col n = lane&15, element i -> K = i + (lane>=16 ? 16 : 0).
// Fragment (kt, nt) stored as 32 lanes x 16 bf16 (1 KB), frag index = kt*NT + nt.
template <int K, int NOUT>
__global__ void pack_w_kernel(const float* __restrict__ W, unsigned short* __restrict__ P) {
  const int NT = NOUT / 16, KT = K / 32;
  int t = blockIdx.x * blockDim.x + threadIdx.x;
  if (t >= KT * NT * 32) return;
  int lane = t & 31;
  int frag = t >> 5;
  int nt = frag % NT;
  int kt = frag / NT;
  int n = nt * 16 + (lane & 15);
  int kbase = kt * 32 + (lane >> 4) * 16;
  unsigned short* o = P + (size_t)t * 16;
#pragma unroll
  for (int i = 0; i < 16; ++i)
    o[i] = f2bf_rne(W[(size_t)(kbase + i) * NOUT + n]);
}

// -------- elementwise f32 -> bf16 (row-major), vectorized --------
__global__ void cvt_bf16_kernel(const float* __restrict__ in,
                                unsigned short* __restrict__ out, long long n4) {
  long long i = (long long)blockIdx.x * blockDim.x + threadIdx.x;
  if (i >= n4) return;
  float4 v = ((const float4*)in)[i];
  ushort4 o;
  o.x = f2bf_rne(v.x); o.y = f2bf_rne(v.y);
  o.z = f2bf_rne(v.z); o.w = f2bf_rne(v.w);
  ((ushort4*)out)[i] = o;
}

// -------- GEMM: C = relu(A1@W1 [+ A2@W2] + bias) --------
// A is bf16 row-major [M x K]. W* are prepacked bf16 B-fragments. One wave computes
// a 16-row x 64-col strip (4 WMMA tiles, A fragment reused 4x). The block's B
// fragments (4 col tiles, all kt, both matrices if DUAL) are staged into LDS with
// async global->LDS copies, then consumed via ds loads.
template <int K, int NOUT, bool DUAL, bool OUT_BF16>
__global__ void gemm_wmma_kernel(const unsigned short* __restrict__ A1,
                                 const unsigned short* __restrict__ W1,
                                 const unsigned short* __restrict__ A2,
                                 const unsigned short* __restrict__ W2,
                                 const float* __restrict__ bias,
                                 void* __restrict__ Cout, int M) {
  const int NT = NOUT / 16;
  const int KT = K / 32;
  const int NG = 4;                    // column tiles per wave
  const int FRAGS = KT * NG;           // fragments per weight matrix staged in LDS
  __shared__ __align__(32) unsigned short smem[(DUAL ? 2 : 1) * FRAGS * 512];

  int lane = threadIdx.x & 31;
  int wave = threadIdx.x >> 5;
  int mTiles = M / 16;
  int nGroups = NT / NG;
  int blockM = blockIdx.x / nGroups;
  int ntg = blockIdx.x % nGroups;

  // ---- async-stage this column group's B fragments into LDS ----
  {
    const int CH = FRAGS * 64;         // 16-byte chunks per matrix
    const int TOT = (DUAL ? 2 : 1) * CH;
    for (int c = threadIdx.x; c < TOT; c += blockDim.x) {
      int m = c / CH;                  // 0: W1, 1: W2
      int cc = c % CH;
      int f = cc >> 6;                 // fragment index within staged set
      int kt = f / NG, ntl = f % NG;
      int ch = cc & 63;                // 16B chunk within fragment
      const unsigned short* src =
          (m ? W2 : W1) + ((size_t)(kt * NT + ntg * NG + ntl)) * 512 + ch * 8;
      unsigned long long ga = (unsigned long long)src;
      unsigned la = (unsigned)(unsigned long long)(void*)
                    &smem[(size_t)m * FRAGS * 512 + f * 512 + ch * 8];
      asm volatile("global_load_async_to_lds_b128 %0, %1, off"
                   :: "v"(la), "v"(ga) : "memory");
    }
    asm volatile("s_wait_asynccnt 0x0" ::: "memory");
    __syncthreads();
  }

  int mTile = blockM * 8 + wave;
  if (mTile >= mTiles) return;         // wave-uniform; after barrier, EXEC stays all-1
  int half = lane >> 4;                // 0: lanes 0-15, 1: lanes 16-31
  int lrow = lane & 15;

  v8f acc[NG] = {};

  // ---- pass 1: A1 @ W1 ----
  {
    const unsigned short* arow = A1 + (size_t)(mTile * 16 + lrow) * K;
#pragma unroll
    for (int kt = 0; kt < KT; ++kt) {
      BFrag a;
      int k0 = kt * 32 + half * 16;
      a.q[0] = *(const uint4*)(arow + k0);        // elements 0-7 : K = k0..k0+7
      a.q[1] = *(const uint4*)(arow + k0 + 8);    // elements 8-15: K = k0+8..k0+15
#pragma unroll
      for (int ntl = 0; ntl < NG; ++ntl) {
        BFrag b;
        b.v = *(const v16bf*)&smem[((kt * NG + ntl) * 32 + lane) * 16];
        acc[ntl] = __builtin_amdgcn_wmma_f32_16x16x32_bf16(
            false, a.v, false, b.v, (short)0, acc[ntl], false, false);
      }
    }
  }

  // ---- pass 2 (fused root term): A2 @ W2 ----
  if constexpr (DUAL) {
    const unsigned short* arow = A2 + (size_t)(mTile * 16 + lrow) * K;
#pragma unroll
    for (int kt = 0; kt < KT; ++kt) {
      BFrag a;
      int k0 = kt * 32 + half * 16;
      a.q[0] = *(const uint4*)(arow + k0);
      a.q[1] = *(const uint4*)(arow + k0 + 8);
#pragma unroll
      for (int ntl = 0; ntl < NG; ++ntl) {
        BFrag b;
        b.v = *(const v16bf*)&smem[(size_t)FRAGS * 512 +
                                   ((kt * NG + ntl) * 32 + lane) * 16];
        acc[ntl] = __builtin_amdgcn_wmma_f32_16x16x32_bf16(
            false, a.v, false, b.v, (short)0, acc[ntl], false, false);
      }
    }
  }

  // ---- epilogue: bias + relu, store f32 or bf16 ----
  float* Cf = (float*)Cout;
  unsigned short* Cb = (unsigned short*)Cout;
#pragma unroll
  for (int ntl = 0; ntl < NG; ++ntl) {
    float bv = bias[(ntg * NG + ntl) * 16 + lrow];
#pragma unroll
    for (int r = 0; r < 8; ++r) {
      float v = fmaxf(acc[ntl][r] + bv, 0.0f);
      // C/D layout: VGPR r -> row r (lanes 0-15) or r+8 (lanes 16-31), col = lane&15
      size_t idx = (size_t)(mTile * 16 + half * 8 + r) * NOUT +
                   (ntg * NG + ntl) * 16 + lrow;
      if constexpr (OUT_BF16) Cb[idx] = f2bf_rne(v);
      else                    Cf[idx] = v;
    }
  }
}

// -------- scatter-max over edges; relu(msgs) >= 0 so u32 bit-order == f32 order --------
template <int D>
__global__ void scatter_max_kernel(const long long* __restrict__ ei,
                                   const float* __restrict__ xp,
                                   unsigned int* __restrict__ aggr) {
  const int D4 = D / 4;
  long long t = (long long)blockIdx.x * blockDim.x + threadIdx.x;
  if (t >= (long long)N_EDGES * D4) return;
  int e = (int)(t / D4);
  int c = (int)(t % D4) * 4;
  int s = (int)ei[e];
  int d = (int)ei[N_EDGES + e];
  float4 v = *(const float4*)(xp + (size_t)s * D + c);
  unsigned int* o = aggr + (size_t)d * D + c;
  atomicMax(o + 0, __float_as_uint(v.x));
  atomicMax(o + 1, __float_as_uint(v.y));
  atomicMax(o + 2, __float_as_uint(v.z));
  atomicMax(o + 3, __float_as_uint(v.w));
}

__global__ void zero_kernel(unsigned int* __restrict__ p, long long n) {
  long long i = (long long)blockIdx.x * blockDim.x + threadIdx.x;
  if (i < n) p[i] = 0u;
}

extern "C" void kernel_launch(void* const* d_in, const int* in_sizes, int n_in,
                              void* d_out, int out_size, void* d_ws, size_t ws_size,
                              hipStream_t stream) {
  const float*     x   = (const float*)d_in[0];
  const long long* ei  = (const long long*)d_in[1];
  const float *Wp1 = (const float*)d_in[2],  *bp1 = (const float*)d_in[3];
  const float *Wl1 = (const float*)d_in[4],  *bl1 = (const float*)d_in[5];
  const float *Wr1 = (const float*)d_in[6];
  const float *Wp2 = (const float*)d_in[7],  *bp2 = (const float*)d_in[8];
  const float *Wl2 = (const float*)d_in[9],  *bl2 = (const float*)d_in[10];
  const float *Wr2 = (const float*)d_in[11];
  float* out = (float*)d_out;

  char* ws = (char*)d_ws;
  size_t cur = 0;
  auto alloc = [&](size_t bytes) -> char* {
    char* p = ws + cur;
    cur += (bytes + 255) & ~(size_t)255;
    return p;
  };

  // packed bf16 weight fragments
  unsigned short* Wp1p = (unsigned short*)alloc((size_t)4 * 8  * 512 * 2);  // 128x128
  unsigned short* Wl1p = (unsigned short*)alloc((size_t)4 * 16 * 512 * 2);  // 128x256
  unsigned short* Wr1p = (unsigned short*)alloc((size_t)4 * 16 * 512 * 2);  // 128x256
  unsigned short* Wp2p = (unsigned short*)alloc((size_t)8 * 16 * 512 * 2);  // 256x256
  unsigned short* Wl2p = (unsigned short*)alloc((size_t)8 * 8  * 512 * 2);  // 256x128
  unsigned short* Wr2p = (unsigned short*)alloc((size_t)8 * 8  * 512 * 2);  // 256x128
  // activations
  unsigned short* xb     = (unsigned short*)alloc((size_t)N_NODES * 128 * 2); // bf16 x
  float*          xproj1 = (float*)alloc((size_t)N_NODES * 128 * 4);
  float*          aggr1  = (float*)alloc((size_t)N_NODES * 128 * 4);
  unsigned short* aggr1b = (unsigned short*)alloc((size_t)N_NODES * 128 * 2);
  unsigned short* h1b    = (unsigned short*)alloc((size_t)N_NODES * 256 * 2); // bf16 h1
  float*          xproj2 = (float*)alloc((size_t)N_NODES * 256 * 4);
  float*          aggr2  = (float*)alloc((size_t)N_NODES * 256 * 4);
  unsigned short* aggr2b = (unsigned short*)alloc((size_t)N_NODES * 256 * 2);

  const int B = 256;
  const int MB = (N_NODES / 16 + 7) / 8;   // row-tile block count (3125 -> 391)

  // ---- pack all weights (tiny) ----
  pack_w_kernel<128, 128><<<(4 * 8  * 32 + B - 1) / B, B, 0, stream>>>(Wp1, Wp1p);
  pack_w_kernel<128, 256><<<(4 * 16 * 32 + B - 1) / B, B, 0, stream>>>(Wl1, Wl1p);
  pack_w_kernel<128, 256><<<(4 * 16 * 32 + B - 1) / B, B, 0, stream>>>(Wr1, Wr1p);
  pack_w_kernel<256, 256><<<(8 * 16 * 32 + B - 1) / B, B, 0, stream>>>(Wp2, Wp2p);
  pack_w_kernel<256, 128><<<(8 * 8  * 32 + B - 1) / B, B, 0, stream>>>(Wl2, Wl2p);
  pack_w_kernel<256, 128><<<(8 * 8  * 32 + B - 1) / B, B, 0, stream>>>(Wr2, Wr2p);

  // x -> bf16 once (feeds gemm1 and gemm2's root term)
  {
    long long n4 = (long long)N_NODES * 128 / 4;
    cvt_bf16_kernel<<<(unsigned)((n4 + B - 1) / B), B, 0, stream>>>(x, xb, n4);
  }

  // ---- layer 1 ----
  // xproj1 = relu(x @ Wp1 + bp1)   [50000 x 128] f32 (scatter needs f32 bits)
  gemm_wmma_kernel<128, 128, false, false><<<MB * 2, B, 0, stream>>>(
      xb, Wp1p, nullptr, nullptr, bp1, xproj1, N_NODES);
  // aggr1 = scatter_max(xproj1[src] -> dst), init 0
  {
    long long n = (long long)N_NODES * 128;
    zero_kernel<<<(unsigned)((n + B - 1) / B), B, 0, stream>>>((unsigned int*)aggr1, n);
    long long w = (long long)N_EDGES * (128 / 4);
    scatter_max_kernel<128><<<(unsigned)((w + B - 1) / B), B, 0, stream>>>(
        ei, xproj1, (unsigned int*)aggr1);
    long long n4 = n / 4;
    cvt_bf16_kernel<<<(unsigned)((n4 + B - 1) / B), B, 0, stream>>>(aggr1, aggr1b, n4);
  }
  // h1 = relu(aggr1 @ Wl1 + bl1 + x @ Wr1)   [50000 x 256], stored bf16 (GEMM-only consumer)
  gemm_wmma_kernel<128, 256, true, true><<<MB * 4, B, 0, stream>>>(
      aggr1b, Wl1p, xb, Wr1p, bl1, h1b, N_NODES);

  // ---- layer 2 ----
  // xproj2 = relu(h1 @ Wp2 + bp2)   [50000 x 256] f32
  gemm_wmma_kernel<256, 256, false, false><<<MB * 4, B, 0, stream>>>(
      h1b, Wp2p, nullptr, nullptr, bp2, xproj2, N_NODES);
  // aggr2 = scatter_max(xproj2[src] -> dst), init 0
  {
    long long n = (long long)N_NODES * 256;
    zero_kernel<<<(unsigned)((n + B - 1) / B), B, 0, stream>>>((unsigned int*)aggr2, n);
    long long w = (long long)N_EDGES * (256 / 4);
    scatter_max_kernel<256><<<(unsigned)((w + B - 1) / B), B, 0, stream>>>(
        ei, xproj2, (unsigned int*)aggr2);
    long long n4 = n / 4;
    cvt_bf16_kernel<<<(unsigned)((n4 + B - 1) / B), B, 0, stream>>>(aggr2, aggr2b, n4);
  }
  // out = relu(aggr2 @ Wl2 + bl2 + h1 @ Wr2)   [50000 x 128] f32 -> d_out
  gemm_wmma_kernel<256, 128, true, false><<<MB * 2, B, 0, stream>>>(
      aggr2b, Wl2p, h1b, Wr2p, bl2, out, N_NODES);
}
